// WSCMAP_10376640987612
// MI455X (gfx1250) — matrix operand order repaired and checked by reading, not compiled, not run on hardware
//
#include <hip/hip_runtime.h>
#include <hip/hip_bf16.h>
#include <math.h>

typedef __attribute__((ext_vector_type(16))) _Float16 v16h;
typedef __attribute__((ext_vector_type(8)))  float    v8f;
typedef __attribute__((ext_vector_type(8)))  _Float16 h8;
// 4-float vector with 4-byte alignment (rows may only be 8B aligned, e.g. ld=50)
typedef float float4a __attribute__((ext_vector_type(4), aligned(4)));

// ---------------------------------------------------------------------------
// Generic batched WMMA GEMM:
//   C[b](m,n) = act( alpha * sum_k A[b](m,k)*B[b](k,n) + bias[n] )
// A: (m,k) at A + boff + m*ldA + k              (row major)
// B: transB ? (k,n) at B + boff + n*ldB + k : B + boff + k*ldB + n
// C: (m,n) at C + boff + m*ldCm + n*ldCn
// batch z: b0 = z/nb1, b1 = z%nb1, boff = b0*s?0 + b1*s?1
// act: 0=none 1=relu 2=sigmoid
//
// Blocking: 128 threads = 4 waves. Block computes a 64(M) x 64(N) macro-tile:
// wave w owns rows [blockIdx.y*64 + w*16, +16) and ALL four 16-wide N tiles.
// A fragment loaded once per wave per k-step, reused by 4 WMMAs.
// Non-transposed B is staged once per block per k-step into 4KB LDS (f16,
// [n][k] layout) by all 128 threads, then read as 16B ds loads by every wave.
// Bounds: M/N overflow only feeds masked-out D rows/cols -> clamp pointers.
// K overflow handled in one masked tail step (cndmask, no divergent branches).
// ---------------------------------------------------------------------------
__global__ __launch_bounds__(128) void gemm_wmma_kernel(
    const float* __restrict__ A, int ldA, long sA0, long sA1,
    const float* __restrict__ B, int ldB, long sB0, long sB1, int transB,
    const float* __restrict__ bias,
    float* __restrict__ C, long ldCm, long ldCn, long sC0, long sC1,
    int M, int N, int K, int nb1, float alpha, int act)
{
  __shared__ __align__(16) _Float16 Bsh[64 * 32];   // [n_local][k] f16 tile, 4KB
  const int b0 = blockIdx.z / nb1, b1 = blockIdx.z % nb1;
  A += (long)b0 * sA0 + (long)b1 * sA1;
  B += (long)b0 * sB0 + (long)b1 * sB1;
  C += (long)b0 * sC0 + (long)b1 * sC1;
  const int wid  = threadIdx.x >> 5;
  const int lane = threadIdx.x & 31;
  const int half = lane >> 4;
  const int l16  = lane & 15;
  const int tm  = (blockIdx.y << 2 | wid) << 4;   // this wave's 16 M rows
  const int tnB = blockIdx.x << 6;                // block's 64-wide N strip
  const int m  = tm + l16;
  const int mc = (m < M) ? m : (M - 1);           // OOB rows masked at store
  const float* __restrict__ Arow = A + (long)mc * ldA;
  // transB: per-N-tile row bases (clamped)
  const float* __restrict__ Btr[4];
  #pragma unroll
  for (int nt = 0; nt < 4; ++nt) {
    int nn = tnB + nt * 16 + l16;
    int nc = (nn < N) ? nn : (N - 1);
    Btr[nt] = B + (long)nc * ldB;
  }
  // staging roles (all 128 threads): row k = srow, 16-col quarter sq
  const int srow = threadIdx.x & 31;
  const int sq   = threadIdx.x >> 5;

  v8f acc[4];
  #pragma unroll
  for (int nt = 0; nt < 4; ++nt) acc[nt] = (v8f){0.f,0.f,0.f,0.f,0.f,0.f,0.f,0.f};
  const int Kfull = K & ~31;

  for (int k0 = 0; k0 < Kfull; k0 += 32) {
    // ---- A fragment (per wave): k = k0+16*half+0..15 ----
    v16h av;
    {
      const float4a a0 = *(const float4a*)(Arow + k0 + 16 * half);
      const float4a a1 = *(const float4a*)(Arow + k0 + 16 * half + 4);
      const float4a a2 = *(const float4a*)(Arow + k0 + 16 * half + 8);
      const float4a a3 = *(const float4a*)(Arow + k0 + 16 * half + 12);
      av[0]=(_Float16)a0.x; av[1]=(_Float16)a0.y; av[2]=(_Float16)a0.z; av[3]=(_Float16)a0.w;
      av[4]=(_Float16)a1.x; av[5]=(_Float16)a1.y; av[6]=(_Float16)a1.z; av[7]=(_Float16)a1.w;
      av[8]=(_Float16)a2.x; av[9]=(_Float16)a2.y; av[10]=(_Float16)a2.z; av[11]=(_Float16)a2.w;
      av[12]=(_Float16)a3.x; av[13]=(_Float16)a3.y; av[14]=(_Float16)a3.z; av[15]=(_Float16)a3.w;
    }
    // ---- B tile staging (non-trans): 128 threads store 32x64 -> LDS ----
    if (!transB) {
      __syncthreads();                           // previous tile fully consumed
      const int kk = k0 + srow;                  // < K in full steps
      const int cb = tnB + sq * 16;
      _Float16 r[16];
      if ((long)kk * ldB + cb + 16 <= (long)K * ldB) {   // row read in-bounds
        const float* brow = B + (long)kk * ldB + cb;
        const float4a c0 = *(const float4a*)(brow);
        const float4a c1 = *(const float4a*)(brow + 4);
        const float4a c2 = *(const float4a*)(brow + 8);
        const float4a c3 = *(const float4a*)(brow + 12);
        r[0]=(_Float16)c0.x; r[1]=(_Float16)c0.y; r[2]=(_Float16)c0.z; r[3]=(_Float16)c0.w;
        r[4]=(_Float16)c1.x; r[5]=(_Float16)c1.y; r[6]=(_Float16)c1.z; r[7]=(_Float16)c1.w;
        r[8]=(_Float16)c2.x; r[9]=(_Float16)c2.y; r[10]=(_Float16)c2.z; r[11]=(_Float16)c2.w;
        r[12]=(_Float16)c3.x; r[13]=(_Float16)c3.y; r[14]=(_Float16)c3.z; r[15]=(_Float16)c3.w;
      } else {                                   // rare edge: clamped scalars
        const long last = (long)K * ldB - 1;
        #pragma unroll
        for (int j = 0; j < 16; ++j) {
          long idx = (long)kk * ldB + cb + j;
          r[j] = (_Float16)B[idx < last ? idx : last];
        }
      }
      #pragma unroll
      for (int j = 0; j < 16; ++j) Bsh[(sq * 16 + j) * 32 + srow] = r[j];
      __syncthreads();
    }
    if (k0 + 32 < K)
      __builtin_prefetch(Arow + k0 + 32, 0, 1);  // global_prefetch_b8
    // ---- 4 WMMAs sharing the A fragment ----
    #pragma unroll
    for (int nt = 0; nt < 4; ++nt) {
      v16h bv;
      if (transB) {
        const float* bt = Btr[nt] + k0 + 16 * half;
        const float4a c0 = *(const float4a*)(bt);
        const float4a c1 = *(const float4a*)(bt + 4);
        const float4a c2 = *(const float4a*)(bt + 8);
        const float4a c3 = *(const float4a*)(bt + 12);
        bv[0]=(_Float16)c0.x; bv[1]=(_Float16)c0.y; bv[2]=(_Float16)c0.z; bv[3]=(_Float16)c0.w;
        bv[4]=(_Float16)c1.x; bv[5]=(_Float16)c1.y; bv[6]=(_Float16)c1.z; bv[7]=(_Float16)c1.w;
        bv[8]=(_Float16)c2.x; bv[9]=(_Float16)c2.y; bv[10]=(_Float16)c2.z; bv[11]=(_Float16)c2.w;
        bv[12]=(_Float16)c3.x; bv[13]=(_Float16)c3.y; bv[14]=(_Float16)c3.z; bv[15]=(_Float16)c3.w;
      } else {
        const h8* bp = (const h8*)(Bsh + (nt * 16 + l16) * 32 + 16 * half);
        h8 lo = bp[0], hi = bp[1];
        #pragma unroll
        for (int j = 0; j < 8; ++j) { bv[j] = lo[j]; bv[8 + j] = hi[j]; }
      }
      acc[nt] = __builtin_amdgcn_wmma_f32_16x16x32_f16(
          false, av, false, bv, (short)0, acc[nt], false, false);
    }
  }

  // ---- K tail (at most one partial step), branch-free masking ----
  if (Kfull < K) {
    const int k0 = Kfull;
    v16h av;
    #pragma unroll
    for (int e = 0; e < 16; ++e) {
      int k = k0 + e + half * 16;
      int kc = (k < K) ? k : (K - 1);
      float x = Arow[kc];
      av[e] = (k < K) ? (_Float16)x : (_Float16)0.f;
    }
    if (!transB) {
      __syncthreads();
      const int kk  = k0 + srow;
      const int kkc = (kk < K) ? kk : (K - 1);
      const int cb  = tnB + sq * 16;
      _Float16 r[16];
      if ((long)kkc * ldB + cb + 16 <= (long)K * ldB) {
        const float* brow = B + (long)kkc * ldB + cb;
        const float4a c0 = *(const float4a*)(brow);
        const float4a c1 = *(const float4a*)(brow + 4);
        const float4a c2 = *(const float4a*)(brow + 8);
        const float4a c3 = *(const float4a*)(brow + 12);
        r[0]=(_Float16)c0.x; r[1]=(_Float16)c0.y; r[2]=(_Float16)c0.z; r[3]=(_Float16)c0.w;
        r[4]=(_Float16)c1.x; r[5]=(_Float16)c1.y; r[6]=(_Float16)c1.z; r[7]=(_Float16)c1.w;
        r[8]=(_Float16)c2.x; r[9]=(_Float16)c2.y; r[10]=(_Float16)c2.z; r[11]=(_Float16)c2.w;
        r[12]=(_Float16)c3.x; r[13]=(_Float16)c3.y; r[14]=(_Float16)c3.z; r[15]=(_Float16)c3.w;
      } else {
        const long last = (long)K * ldB - 1;
        #pragma unroll
        for (int j = 0; j < 16; ++j) {
          long idx = (long)kkc * ldB + cb + j;
          r[j] = (_Float16)B[idx < last ? idx : last];
        }
      }
      const _Float16 zero = (_Float16)0.f;
      #pragma unroll
      for (int j = 0; j < 16; ++j)
        Bsh[(sq * 16 + j) * 32 + srow] = (kk < K) ? r[j] : zero;
      __syncthreads();
    }
    #pragma unroll
    for (int nt = 0; nt < 4; ++nt) {
      v16h bv;
      if (transB) {
        #pragma unroll
        for (int e = 0; e < 16; ++e) {
          int k = k0 + e + half * 16;
          int kc = (k < K) ? k : (K - 1);
          float x = Btr[nt][kc];
          bv[e] = (k < K) ? (_Float16)x : (_Float16)0.f;
        }
      } else {
        const h8* bp = (const h8*)(Bsh + (nt * 16 + l16) * 32 + 16 * half);
        h8 lo = bp[0], hi = bp[1];
        #pragma unroll
        for (int j = 0; j < 8; ++j) { bv[j] = lo[j]; bv[8 + j] = hi[j]; }
      }
      acc[nt] = __builtin_amdgcn_wmma_f32_16x16x32_f16(
          false, av, false, bv, (short)0, acc[nt], false, false);
    }
  }

  // ---- epilogue: C/D layout VGPR r -> M = r + 8*half, N = lane&15 ----
  #pragma unroll
  for (int nt = 0; nt < 4; ++nt) {
    #pragma unroll
    for (int r = 0; r < 8; ++r) {
      int mm = tm + r + half * 8;
      int nn = tnB + nt * 16 + l16;
      if (mm < M && nn < N) {
        float v = acc[nt][r] * alpha;
        if (bias) v += bias[nn];
        if (act == 1) v = fmaxf(v, 0.f);
        else if (act == 2) v = 1.f / (1.f + __expf(-v));
        C[(long)mm * ldCm + (long)nn * ldCn] = v;
      }
    }
  }
}

// ---------------------------------------------------------------------------
// TDM arity probe (NOT launched): verifies __builtin_amdgcn_tensor_load_to_lds
// signature on this toolchain (compiled OK in round 3).
// ---------------------------------------------------------------------------
typedef unsigned int u32x4 __attribute__((ext_vector_type(4)));
typedef int          i32x4 __attribute__((ext_vector_type(4)));
typedef int          i32x8 __attribute__((ext_vector_type(8)));

__global__ void tdm_probe_kernel(unsigned int* flag) {
  __shared__ float lds[256];
  u32x4 g0 = {};
  i32x8 g1 = {};
  i32x4 g2 = {}, g3 = {};
#if __has_include(<hip/amd_detail/amd_gfx1250_TDM.h>)
  i32x8 g1b = {};
  __builtin_amdgcn_tensor_load_to_lds(g0, g1, g2, g3, g1b, 0);
#else
  __builtin_amdgcn_tensor_load_to_lds(g0, g1, g2, g3, 0);
#endif
  __builtin_amdgcn_s_wait_tensorcnt(0);
  if (threadIdx.x == 0) flag[0] = (unsigned)lds[0];
}

// ---------------------------------------------------------------------------
// LayerNorm over last dim (<=128), one wave per row. y = LN(x (+res)) * g + b
// ---------------------------------------------------------------------------
__global__ __launch_bounds__(32) void ln_kernel(
    const float* __restrict__ x, const float* __restrict__ res,
    const float* __restrict__ g, const float* __restrict__ b,
    float* __restrict__ y, int dim)
{
  long row = blockIdx.x;
  const float* xr = x + row * (long)dim;
  const float* rr = res ? res + row * (long)dim : nullptr;
  float* yr = y + row * (long)dim;
  int lane = threadIdx.x;
  float v[4];
  float s = 0.f;
  #pragma unroll
  for (int i = 0; i < 4; ++i) {
    int c = lane + i * 32;
    float t = 0.f;
    if (c < dim) { t = xr[c]; if (rr) t += rr[c]; }
    v[i] = t; s += t;
  }
  #pragma unroll
  for (int off = 16; off > 0; off >>= 1) s += __shfl_xor(s, off, 32);
  float mean = s / (float)dim;
  float q = 0.f;
  #pragma unroll
  for (int i = 0; i < 4; ++i) {
    int c = lane + i * 32;
    if (c < dim) { float d = v[i] - mean; q += d * d; }
  }
  #pragma unroll
  for (int off = 16; off > 0; off >>= 1) q += __shfl_xor(q, off, 32);
  float rstd = rsqrtf(q / (float)dim + 1e-5f);
  #pragma unroll
  for (int i = 0; i < 4; ++i) {
    int c = lane + i * 32;
    if (c < dim) yr[c] = (v[i] - mean) * rstd * g[c] + b[c];
  }
}

// ---------------------------------------------------------------------------
// Softmax over rows of length cols (<=192), one wave per row.
// Optional pair bias (triangle attention): row = ((i_local*nh)+h)*qlen + q,
// bias value for col k is bias[((i0+i_local)*cols + k)*nh + h].
// ---------------------------------------------------------------------------
__global__ __launch_bounds__(32) void softmax_kernel(
    float* __restrict__ s, int cols,
    const float* __restrict__ bias, int i0, int nh, int qlen)
{
  long row = blockIdx.x;
  float* sr = s + row * (long)cols;
  const float* bp = nullptr;
  if (bias) {
    long il = row / ((long)nh * qlen);
    long h  = (row / qlen) % nh;
    bp = bias + ((long)(i0 + il) * cols) * nh + h;     // element k at bp[k*nh]
  }
  int lane = threadIdx.x;
  float v[6];
  float mx = -1e30f;
  #pragma unroll
  for (int i = 0; i < 6; ++i) {
    int c = lane + i * 32;
    float t = -1e30f;
    if (c < cols) { t = sr[c]; if (bp) t += bp[(long)c * nh]; }
    v[i] = t; mx = fmaxf(mx, t);
  }
  #pragma unroll
  for (int off = 16; off > 0; off >>= 1) mx = fmaxf(mx, __shfl_xor(mx, off, 32));
  float sum = 0.f;
  #pragma unroll
  for (int i = 0; i < 6; ++i) {
    int c = lane + i * 32;
    if (c < cols) { v[i] = __expf(v[i] - mx); sum += v[i]; } else v[i] = 0.f;
  }
  #pragma unroll
  for (int off = 16; off > 0; off >>= 1) sum += __shfl_xor(sum, off, 32);
  float inv = 1.f / sum;
  #pragma unroll
  for (int i = 0; i < 6; ++i) {
    int c = lane + i * 32;
    if (c < cols) sr[c] = v[i] * inv;
  }
}

// ---------------------------------------------------------------------------
// Elementwise helpers
// ---------------------------------------------------------------------------
__global__ void add_pe_kernel(float* __restrict__ x, long n, int dim) {
  long i = (long)blockIdx.x * blockDim.x + threadIdx.x;
  if (i < n) x[i] += (float)((i % dim) & 1);   // pe0 = [0,1,0,1,...]
}
__global__ void sig_add_kernel(float* __restrict__ x, const float* __restrict__ z, long n) {
  long i = (long)blockIdx.x * blockDim.x + threadIdx.x;
  if (i < n) x[i] += 1.f / (1.f + __expf(-z[i]));
}
__global__ void sigmoid_ip_kernel(float* __restrict__ x, long n) {
  long i = (long)blockIdx.x * blockDim.x + threadIdx.x;
  if (i < n) x[i] = 1.f / (1.f + __expf(-x[i]));
}
__global__ void add_kernel(float* __restrict__ a, const float* __restrict__ b, long n) {
  long i = (long)blockIdx.x * blockDim.x + threadIdx.x;
  if (i < n) a[i] += b[i];
}
// z [192,192,50] -> zt [192,192,50] with i/j swapped
__global__ void transpose_ij_kernel(const float* __restrict__ src, float* __restrict__ dst) {
  long idx = (long)blockIdx.x * blockDim.x + threadIdx.x;
  if (idx >= 36864L * 50) return;
  long ij = idx / 50; int c = (int)(idx % 50);
  long i = ij / 192, j = ij % 192;
  dst[(j * 192 + i) * 50 + c] = src[idx];
}
// o[p,hc] *= sigmoid(zn[base+p,:] . Wg[:,hc] + bg[hc])   (50-long dot)
__global__ void gate_kernel(float* __restrict__ o, const float* __restrict__ zn,
                            const float* __restrict__ Wg, const float* __restrict__ bg,
                            long npos, long base) {
  long idx = (long)blockIdx.x * blockDim.x + threadIdx.x;
  if (idx >= npos * 500) return;
  long p = idx / 500; int hc = (int)(idx % 500);
  const float* zr = zn + (base + p) * 50;
  float acc = bg[hc];
  #pragma unroll
  for (int c = 0; c < 50; ++c) acc += zr[c] * Wg[(long)c * 500 + hc];
  o[idx] *= 1.f / (1.f + __expf(-acc));
}
// t [36864,25] -> out [25,36864] with triangular bin smoothing [.25,.5,.25]
__global__ void smooth_kernel(const float* __restrict__ t, float* __restrict__ out) {
  long idx = (long)blockIdx.x * blockDim.x + threadIdx.x;
  if (idx >= 25L * 36864) return;
  int o = (int)(idx / 36864); long ij = idx % 36864;
  const float* tr = t + ij * 25;
  float v = 0.5f * tr[o];
  if (o > 0)  v += 0.25f * tr[o - 1];
  if (o < 24) v += 0.25f * tr[o + 1];
  out[idx] = v;
}

// ---------------------------------------------------------------------------
// Host-side parameter bundles (jax pytree order: dict keys sorted)
// ---------------------------------------------------------------------------
struct AttnP { const float *Wk,*Wo,*Wq,*Wv,*bk,*bo,*bq,*bv; };
struct DecP  { const float *W1,*W2,*b1,*b2; AttnP ca;
               const float *ln1b,*ln1g,*ln2b,*ln2g,*ln3b,*ln3g; AttnP sa; };
struct EncP  { const float *W1,*W2,*b1,*b2,*ln1b,*ln1g,*ln2b,*ln2g; AttnP sa; };
struct TaP   { const float *Wb,*Wg,*Wk,*Wo,*Wq,*Wv,*bg,*bo,*lnb,*lng; };

extern "C" void kernel_launch(void* const* d_in, const int* in_sizes, int n_in,
                              void* d_out, int out_size, void* d_ws, size_t ws_size,
                              hipStream_t stream) {
  const int L = 192, D = 100, HID = 256, INIT = 1280, CB = 50, NH = 5, CH = 100;
  const long LD = (long)L * D, LL = (long)L * L;          // 19200, 36864
  const int CHUNK = 48, NCHUNK = 4;                        // triangle i-chunks

  int pi = 0;
  auto in = [&](void) { return (const float*)d_in[pi++]; };
  const float* x1in = in();
  const float* x2in = in();
  const float* W_bil = in();                               // params['W'] [50,100,50]
  auto readAttn = [&](AttnP& a) { a.Wk=in(); a.Wo=in(); a.Wq=in(); a.Wv=in();
                                  a.bk=in(); a.bo=in(); a.bq=in(); a.bv=in(); };
  DecP dec[2];
  for (int l = 0; l < 2; ++l) {
    DecP& d = dec[l];
    d.W1=in(); d.W2=in(); d.b1=in(); d.b2=in();
    readAttn(d.ca);
    d.ln1b=in(); d.ln1g=in(); d.ln2b=in(); d.ln2g=in(); d.ln3b=in(); d.ln3g=in();
    readAttn(d.sa);
  }
  EncP enc[2];
  for (int l = 0; l < 2; ++l) {
    EncP& e = enc[l];
    e.W1=in(); e.W2=in(); e.b1=in(); e.b2=in();
    e.ln1b=in(); e.ln1g=in(); e.ln2b=in(); e.ln2g=in();
    readAttn(e.sa);
  }
  const float* out_W = in();
  const float* out_b = in();
  const float* pW1 = in();
  const float* pW2 = in();
  const float* pb1 = in();
  const float* pb2 = in();
  TaP tri[4][2];                                            // [t][0]=end, [t][1]=start
  for (int t = 0; t < 4; ++t)
    for (int w = 0; w < 2; ++w) {
      TaP& a = tri[t][w];
      a.Wb=in(); a.Wg=in(); a.Wk=in(); a.Wo=in(); a.Wq=in(); a.Wv=in();
      a.bg=in(); a.bo=in(); a.lnb=in(); a.lng=in();
    }

  // ---- workspace bump allocator (floats) ----
  size_t off = 0;
  auto wal = [&](size_t n) { float* p = (float*)d_ws + off; off += n; return p; };
  float* X1 = wal(LD);  float* X2 = wal(LD);
  float* Y1 = wal(LD);  float* Y2 = wal(LD);
  float* ZD1 = wal(LD); float* ZD2 = wal(LD);
  float* T256 = wal((size_t)L * HID);
  float* QB = wal(LD);  float* KB = wal(LD);  float* VB = wal(LD);
  float* S  = wal((size_t)NH * LL);
  float* AO = wal(LD);  float* TMP = wal(LD); float* H1 = wal(LD);
  float* U1 = wal((size_t)CB * L * CB); float* U2 = wal((size_t)CB * L * CB);
  float* Z  = wal(LL * CB); float* ZT = wal(LL * CB); float* ZN = wal(LL * CB);
  float* BIASB = wal(LL * NH);
  float* QC = wal((size_t)CHUNK * L * NH * CH);
  float* KC = wal((size_t)CHUNK * L * NH * CH);
  float* VC = wal((size_t)CHUNK * L * NH * CH);
  float* OC = wal((size_t)CHUNK * L * NH * CH);
  float* SC = wal((size_t)CHUNK * NH * LL);
  float* PC = wal((size_t)CHUNK * L * CB);
  float* OUTP = wal(LL * 25);
  (void)ws_size; (void)n_in; (void)in_sizes; (void)out_size;

  auto gemm = [&](const float* A, int ldA, long sA0, long sA1,
                  const float* B, int ldB, long sB0, long sB1, int transB,
                  const float* bias, float* C, long ldCm, long ldCn, long sC0, long sC1,
                  int M, int N, int K, int batches, int nb1, float alpha, int act) {
    dim3 g((N + 63) / 64, (M + 63) / 64, batches);   // 64x64 macro-tile per block
    gemm_wmma_kernel<<<g, 128, 0, stream>>>(A, ldA, sA0, sA1, B, ldB, sB0, sB1, transB,
                                            bias, C, ldCm, ldCn, sC0, sC1,
                                            M, N, K, nb1, alpha, act);
  };
  auto ln = [&](const float* x, const float* res, const float* g, const float* b,
                float* y, int rows, int dim) {
    ln_kernel<<<rows, 32, 0, stream>>>(x, res, g, b, y, dim);
  };
  auto ew = [&](long n) { return dim3((unsigned)((n + 255) / 256)); };

  const float rdh = 0.22360679774997896f;   // 1/sqrt(20)
  // MHA (PyTorch MultiheadAttention, D=100, H=5, dh=20) -> out
  auto mha = [&](const float* qsrc, const float* kvsrc, const AttnP& a, float* out) {
    gemm(qsrc, D,0,0,  a.Wq, D,0,0,0, a.bq, QB, D,1,0,0, L,D,D, 1,1, 1.f, 0);
    gemm(kvsrc,D,0,0,  a.Wk, D,0,0,0, a.bk, KB, D,1,0,0, L,D,D, 1,1, 1.f, 0);
    gemm(kvsrc,D,0,0,  a.Wv, D,0,0,0, a.bv, VB, D,1,0,0, L,D,D, 1,1, 1.f, 0);
    // scores per head: S_h = (Q_h K_h^T)/sqrt(dh)
    gemm(QB, D,0,20,  KB, D,0,20,1, nullptr, S, L,1, 0, LL, L,L,20, NH,NH, rdh, 0);
    softmax_kernel<<<NH * L, 32, 0, stream>>>(S, L, nullptr, 0, 1, L);
    // O_h = A_h V_h
    gemm(S, L,0,LL,  VB, D,0,20,0, nullptr, AO, D,1, 0,20, L,20,L, NH,NH, 1.f, 0);
    gemm(AO, D,0,0,  a.Wo, D,0,0,0, a.bo, out, D,1,0,0, L,D,D, 1,1, 1.f, 0);
  };
  auto enc_layer = [&](float* Y, const EncP& e) {
    mha(Y, Y, e.sa, TMP);
    ln(Y, TMP, e.ln1g, e.ln1b, Y, L, D);
    gemm(Y, D,0,0, e.W1, D,0,0,0, e.b1, H1, D,1,0,0, L,D,D, 1,1, 1.f, 1);
    gemm(H1,D,0,0, e.W2, D,0,0,0, e.b2, TMP,D,1,0,0, L,D,D, 1,1, 1.f, 0);
    ln(Y, TMP, e.ln2g, e.ln2b, Y, L, D);
  };
  auto dec_layer = [&](float* Zb, const float* mem, const DecP& d) {
    mha(Zb, Zb, d.sa, TMP);
    ln(Zb, TMP, d.ln1g, d.ln1b, Zb, L, D);
    mha(Zb, mem, d.ca, TMP);
    ln(Zb, TMP, d.ln2g, d.ln2b, Zb, L, D);
    gemm(Zb, D,0,0, d.W1, D,0,0,0, d.b1, H1, D,1,0,0, L,D,D, 1,1, 1.f, 1);
    gemm(H1, D,0,0, d.W2, D,0,0,0, d.b2, TMP,D,1,0,0, L,D,D, 1,1, 1.f, 0);
    ln(Zb, TMP, d.ln3g, d.ln3b, Zb, L, D);
  };
  // Triangle attention over rows, residual added into Zb
  auto tri_att = [&](float* Zb, const TaP& t) {
    ln(Zb, nullptr, t.lng, t.lnb, ZN, (int)LL, CB);
    gemm(ZN, CB,0,0, t.Wb, NH,0,0,0, nullptr, BIASB, NH,1,0,0, (int)LL, NH, CB, 1,1, 1.f, 0);
    for (int ci = 0; ci < NCHUNK; ++ci) {
      const int i0 = ci * CHUNK;
      const float* znc = ZN + (long)i0 * L * CB;
      const int MP = CHUNK * L;                             // 9216 positions
      gemm(znc, CB,0,0, t.Wq, NH*CH,0,0,0, nullptr, QC, NH*CH,1,0,0, MP, NH*CH, CB, 1,1, 0.1f, 0);
      gemm(znc, CB,0,0, t.Wk, NH*CH,0,0,0, nullptr, KC, NH*CH,1,0,0, MP, NH*CH, CB, 1,1, 1.f, 0);
      gemm(znc, CB,0,0, t.Wv, NH*CH,0,0,0, nullptr, VC, NH*CH,1,0,0, MP, NH*CH, CB, 1,1, 1.f, 0);
      // scores: per (i_local,h): [192,192] = q [192,100] * k^T
      gemm(QC, NH*CH, (long)L*NH*CH, CH,
           KC, NH*CH, (long)L*NH*CH, CH, 1, nullptr,
           SC, L, 1, (long)NH*LL, LL,
           L, L, CH, CHUNK*NH, NH, 1.f, 0);
      softmax_kernel<<<CHUNK * NH * L, 32, 0, stream>>>(SC, L, BIASB, i0, NH, L);
      // O = A * V
      gemm(SC, L, (long)NH*LL, LL,
           VC, NH*CH, (long)L*NH*CH, CH, 0, nullptr,
           OC, NH*CH, 1, (long)L*NH*CH, CH,
           L, CH, L, CHUNK*NH, NH, 1.f, 0);
      gate_kernel<<<ew((long)MP * NH * CH), 256, 0, stream>>>(OC, ZN, t.Wg, t.bg,
                                                              MP, (long)i0 * L);
      gemm(OC, NH*CH,0,0, t.Wo, CB,0,0,0, t.bo, PC, CB,1,0,0, MP, CB, NH*CH, 1,1, 1.f, 0);
      add_kernel<<<ew((long)MP * CB), 256, 0, stream>>>(Zb + (long)i0 * L * CB, PC, (long)MP * CB);
    }
  };

  // ================= forward =================
  // projection head: relu(relu(x @ pW1 + pb1) @ pW2 + pb2) + pe0
  gemm(x1in, INIT,0,0, pW1, HID,0,0,0, pb1, T256, HID,1,0,0, L,HID,INIT, 1,1, 1.f, 1);
  gemm(T256, HID,0,0,  pW2, D,0,0,0,  pb2, X1,   D,1,0,0,   L,D,HID,    1,1, 1.f, 1);
  add_pe_kernel<<<ew(LD), 256, 0, stream>>>(X1, LD, D);
  gemm(x2in, INIT,0,0, pW1, HID,0,0,0, pb1, T256, HID,1,0,0, L,HID,INIT, 1,1, 1.f, 1);
  gemm(T256, HID,0,0,  pW2, D,0,0,0,  pb2, X2,   D,1,0,0,   L,D,HID,    1,1, 1.f, 1);
  add_pe_kernel<<<ew(LD), 256, 0, stream>>>(X2, LD, D);

  // shared encoder on both sequences
  hipMemcpyAsync(Y1, X1, LD * sizeof(float), hipMemcpyDeviceToDevice, stream);
  hipMemcpyAsync(Y2, X2, LD * sizeof(float), hipMemcpyDeviceToDevice, stream);
  for (int l = 0; l < 2; ++l) { enc_layer(Y1, enc[l]); enc_layer(Y2, enc[l]); }

  // shared decoder, swapped memory
  hipMemcpyAsync(ZD1, Y1, LD * sizeof(float), hipMemcpyDeviceToDevice, stream);
  hipMemcpyAsync(ZD2, Y2, LD * sizeof(float), hipMemcpyDeviceToDevice, stream);
  for (int l = 0; l < 2; ++l) dec_layer(ZD1, Y2, dec[l]);
  for (int l = 0; l < 2; ++l) dec_layer(ZD2, Y1, dec[l]);

  sig_add_kernel<<<ew(LD), 256, 0, stream>>>(X1, ZD1, LD);
  sig_add_kernel<<<ew(LD), 256, 0, stream>>>(X2, ZD2, LD);
  sigmoid_ip_kernel<<<ew(LD), 256, 0, stream>>>(X1, LD);
  sigmoid_ip_kernel<<<ew(LD), 256, 0, stream>>>(X2, LD);

  // bilinear contact map: u[k] = sigmoid(x @ W[k]), cm[k] = u1[k] u2[k]^T -> z[i,j,k]
  gemm(X1, D,0,0, W_bil, CB,(long)D*CB,0,0, nullptr, U1, CB,1,(long)L*CB,0,
       L, CB, D, CB,1, 1.f, 2);
  gemm(X2, D,0,0, W_bil, CB,(long)D*CB,0,0, nullptr, U2, CB,1,(long)L*CB,0,
       L, CB, D, CB,1, 1.f, 2);
  gemm(U1, CB,(long)L*CB,0, U2, CB,(long)L*CB,0,1, nullptr,
       Z, (long)L*CB, CB, 1, 0, L, L, CB, CB,1, 1.f, 0);

  // 4 triangle blocks: start on rows, end on columns (via transpose)
  for (int t = 0; t < 4; ++t) {
    tri_att(Z, tri[t][1]);                                  // 'start'
    transpose_ij_kernel<<<ew(LL * CB), 256, 0, stream>>>(Z, ZT);
    tri_att(ZT, tri[t][0]);                                 // 'end'
    transpose_ij_kernel<<<ew(LL * CB), 256, 0, stream>>>(ZT, Z);
  }

  // output bins + sigmoid, then triangular smoothing over bins
  gemm(Z, CB,0,0, out_W, 25,0,0,0, out_b, OUTP, 25,1,0,0, (int)LL, 25, CB, 1,1, 1.f, 2);
  smooth_kernel<<<ew(25L * LL), 256, 0, stream>>>(OUTP, (float*)d_out);
}
